// Model_66236985639823
// MI455X (gfx1250) — matrix operand order repaired and checked by reading, not compiled, not run on hardware
//
#include <hip/hip_runtime.h>

typedef __bf16 bf16_t;
typedef bf16_t v16bf  __attribute__((ext_vector_type(16)));
typedef bf16_t bf16x8 __attribute__((ext_vector_type(8)));
typedef float  v8f    __attribute__((ext_vector_type(8)));

namespace {

constexpr int Bb = 16, Ll = 336, Hh = 192, Cc = 862, Kk = 8, Ssz = 8;
constexpr int FREQ = 97;                 // H/2+1
constexpr int ODIM = FREQ * 17;          // 1649
constexpr int BC   = Bb * Cc;            // 13792
constexpr int BCK  = BC * Kk;            // 110336
constexpr int NB   = 256;                // padded operator width (columns)
constexpr int MIXK = 608;                // 2*291 padded to mult of 32
constexpr int IRK  = 224;                // 2*97 padded to mult of 32

__device__ __forceinline__ bf16_t f2b(float f) {
  union { float f; unsigned u; } in; in.f = f;
  unsigned u = in.u;
  unsigned r = u + 0x7FFFu + ((u >> 16) & 1u);   // round-to-nearest-even
  union { unsigned short s; bf16_t b; } out;
  out.s = (unsigned short)(r >> 16);
  return out.b;
}

// ---------------------------------------------------------------------------
// Operator-matrix builders.  Stored COLUMN-MAJOR (Bsw[n*KD + k]) so a WMMA
// B-fragment (lane = column n, K = khi..khi+15) is two contiguous 16B loads.
// ---------------------------------------------------------------------------
__global__ __launch_bounds__(256) void init_dft(bf16_t* D) {
  int t = blockIdx.x;           // 0..191 (time = K dim)
  int col = threadIdx.x;        // 0..255
  float v = 0.f;
  if (col < FREQ) {
    int m = (t * col) % Hh;
    v = cosf(6.283185307179586f * (float)m / (float)Hh);
  } else if (col < 2 * FREQ) {
    int f = col - FREQ;
    int m = (t * f) % Hh;
    v = -sinf(6.283185307179586f * (float)m / (float)Hh);
  }
  D[(size_t)col * Hh + t] = f2b(v);
}

__global__ __launch_bounds__(256) void init_ir(bf16_t* G) {
  int j = blockIdx.x;           // 0..223 (spectral row = K dim: 0..96 Re, 97..193 Im)
  int n = threadIdx.x;          // 0..255 (time col, valid < 192)
  float v = 0.f;
  if (n < Hh && j < 2 * FREQ) {
    int f = (j < FREQ) ? j : j - FREQ;
    float cf = (f == 0 || f == FREQ - 1) ? 1.f : 2.f;
    int m = (f * n) % Hh;
    float ang = 6.283185307179586f * (float)m / (float)Hh;
    v = ((j < FREQ) ? cosf(ang) : -sinf(ang)) * cf / (float)Hh;
  }
  G[(size_t)n * IRK + j] = f2b(v);
}

__global__ __launch_bounds__(256) void init_mix(bf16_t* M, const float* wr, const float* wi) {
  int j = blockIdx.x;           // 0..607 (input = K dim: 0..290 Re, 291..581 Im)
  int o2 = threadIdx.x;         // 0..255 (output col: 0..96 Re, 97..193 Im)
  float v = 0.f;
  if (j < 582 && o2 < 2 * FREQ) {
    if (o2 < FREQ) {            // Re(out) = Re*wr - Im*wi
      int o = o2;
      v = (j < 291) ? wr[o * 291 + j] : -wi[o * 291 + (j - 291)];
    } else {                    // Im(out) = Re*wi + Im*wr
      int o = o2 - FREQ;
      v = (j < 291) ? wi[o * 291 + j] : wr[o * 291 + (j - 291)];
    }
  }
  M[(size_t)o2 * MIXK + j] = f2b(v);
}

// ---------------------------------------------------------------------------
// k1: per-channel stats, classifier softmax p, corr softmax, sign(r),
//     yn (f32 + bf16), xn tail (last 48). One lane per channel (coalesced).
// ---------------------------------------------------------------------------
__global__ __launch_bounds__(256) void k1_stats(
    const float* __restrict__ x, const float* __restrict__ yh,
    const float* __restrict__ r_in, const float* __restrict__ temp,
    const float* __restrict__ cw, const float* __restrict__ bst,
    const float* __restrict__ sbias,
    float* __restrict__ stdv, float* __restrict__ pC,
    float* __restrict__ corr, float* __restrict__ sgn,
    float* __restrict__ yn, bf16_t* __restrict__ ynb,
    float* __restrict__ xtail) {
  __shared__ float scw[Ssz * Ll];
  int tid = threadIdx.x;
  for (int i = tid; i < Ssz * Ll; i += 256) scw[i] = cw[i];
  __syncthreads();

  int b = blockIdx.y;
  int c = blockIdx.x * 256 + tid;
  bool active = (c < Cc);

  float s0 = 0.f, s2 = 0.f;
  float acc[Ssz];
#pragma unroll
  for (int s = 0; s < Ssz; ++s) acc[s] = 0.f;

  for (int l = 0; l < Ll; ++l) {
    float v = active ? x[((size_t)b * Ll + l) * Cc + c] : 0.f;
    s0 += v; s2 += v * v;
#pragma unroll
    for (int s = 0; s < Ssz; ++s) acc[s] += v * scw[s * Ll + l];
  }
  float mu = s0 / (float)Ll;
  float var = s2 / (float)Ll - mu * mu;
  float sd = sqrtf(var + 1e-8f);

  if (!active) return;
  int bc = b * Cc + c;
  stdv[bc] = sd;

  float lg[Ssz], mx = -1e30f;
#pragma unroll
  for (int s = 0; s < Ssz; ++s) {
    lg[s] = sbias[s] + bst[c * Ssz + s] + acc[s];
    mx = fmaxf(mx, lg[s]);
  }
  float den = 0.f;
#pragma unroll
  for (int s = 0; s < Ssz; ++s) { lg[s] = expf(lg[s] - mx); den += lg[s]; }
#pragma unroll
  for (int s = 0; s < Ssz; ++s) pC[bc * Ssz + s] = lg[s] / den;

  float T = temp[0];
  float ra[Kk];
  float m2 = 1.f / T;
#pragma unroll
  for (int k = 0; k < Kk; ++k) {
    float rv = r_in[bc * Kk + k];
    sgn[bc * Kk + k] = (rv > 0.f) ? 1.f : ((rv < 0.f) ? -1.f : 0.f);
    ra[k] = fabsf(rv) / T;
    m2 = fmaxf(m2, ra[k]);
  }
  float den2 = expf(1.f / T - m2);
#pragma unroll
  for (int k = 0; k < Kk; ++k) { ra[k] = expf(ra[k] - m2); den2 += ra[k]; }
#pragma unroll
  for (int k = 0; k < Kk; ++k) corr[bc * Kk + k] = ra[k] / den2;

  for (int t = 0; t < 48; ++t) {
    float v = x[((size_t)b * Ll + 288 + t) * Cc + c];
    xtail[(size_t)bc * 48 + t] = (v - mu) / sd;
  }
  for (int h = 0; h < Hh; ++h) {
    float v = yh[((size_t)b * Hh + h) * Cc + c];
    float w = (v - mu) / sd;
    yn[(size_t)bc * Hh + h] = w;
    ynb[(size_t)bc * Hh + h] = f2b(w);
  }
}

// ---------------------------------------------------------------------------
// k2: gather shifted leader sequences -> Sm (BCK x 192 bf16)
// ---------------------------------------------------------------------------
__global__ __launch_bounds__(Hh) void k2_gather(
    const int* __restrict__ leader, const int* __restrict__ shift,
    const float* __restrict__ yn, const float* __restrict__ xtail,
    bf16_t* __restrict__ Sm) {
  int r = blockIdx.x;                 // bc*K + k
  int bc = r >> 3;
  int b = bc / Cc;
  int l = leader[r];
  int sh = shift[r];
  size_t src = (size_t)(b * Cc + l);
  int t = threadIdx.x;
  int u = t - sh;
  float v = (u >= 0) ? yn[src * Hh + u] : xtail[src * 48 + 48 + u];
  Sm[(size_t)r * Hh + t] = f2b(v);
}

// ---------------------------------------------------------------------------
// WMMA GEMM: C(MxN) = A(MxKD,bf16,row-major,lda) x Bsw(col-major: [n*KD+k])
// block = 256 thr = 8 waves -> 32(M) x 128(N); per wave 16x32 (2 WMMAs).
// A tile staged in LDS via global_load_async_to_lds_b64 (ASYNCcnt).
// A fragment = 2x ds_load_b128, B fragment = 2x global_load_b128 (L2-resident).
// MODE 0: f32 out.  MODE 1: + complex bias -> bf16 (zero-pad cols 194..223).
// MODE 2: out = y_hat + val*std -> d_out laid out (B,H,C).
// ---------------------------------------------------------------------------
template <int MODE, int KD>
__global__ __launch_bounds__(256) void gemm_wmma(
    const bf16_t* __restrict__ A, const bf16_t* __restrict__ Bsw,
    float* __restrict__ Cout, bf16_t* __restrict__ Cb16,
    const float* __restrict__ bias_re, const float* __restrict__ bias_im,
    const float* __restrict__ yhat, const float* __restrict__ stdv,
    float* __restrict__ outp,
    int M, int lda, int ldc, int Nvalid) {
  __shared__ alignas(16) bf16_t sA[32 * 32];

  int tid  = threadIdx.x;
  int lane = tid & 31;
  int wid  = tid >> 5;
  int m_off = (wid & 1) * 16;
  int n_off = (wid >> 1) * 32;
  int row0 = blockIdx.x * 32;
  int col0 = blockIdx.y * 128;

  // async A staging: 256 thr x 8B = full 32x32 bf16 tile
  int ar = tid >> 3;            // row in tile
  int ac = (tid & 7) * 4;       // col (k) in tile
  unsigned ldsoff = (unsigned)(unsigned long long)(const void*)sA + (unsigned)(tid * 8);
  const bf16_t* agbase = A + (size_t)(row0 + ar) * lda + ac;

  int mrow  = m_off + (lane & 15);
  int khalf = (lane >= 16) ? 16 : 0;    // A: K = khalf + i
  int khi   = (lane >= 16) ? 16 : 0;    // B: K = khi + i
  const bf16x8* pb0 =
      (const bf16x8*)(Bsw + (size_t)(col0 + n_off + (lane & 15)) * KD + khi);
  const bf16x8* pb1 =
      (const bf16x8*)(Bsw + (size_t)(col0 + n_off + 16 + (lane & 15)) * KD + khi);

  v8f acc0 = {0.f, 0.f, 0.f, 0.f, 0.f, 0.f, 0.f, 0.f};
  v8f acc1 = {0.f, 0.f, 0.f, 0.f, 0.f, 0.f, 0.f, 0.f};

  for (int kb = 0; kb < KD; kb += 32) {
    unsigned long long ga = (unsigned long long)(agbase + kb);
    asm volatile("global_load_async_to_lds_b64 %0, %1, off"
                 :: "v"(ldsoff), "v"(ga) : "memory");
    asm volatile("s_wait_asynccnt 0" ::: "memory");
    __syncthreads();

    const bf16x8* pa = (const bf16x8*)(sA + mrow * 32 + khalf);
    bf16x8 a_lo = pa[0];          // K = khalf..khalf+7
    bf16x8 a_hi = pa[1];          // K = khalf+8..khalf+15
    v16bf fa = __builtin_shufflevector(a_lo, a_hi,
        0, 1, 2, 3, 4, 5, 6, 7, 8, 9, 10, 11, 12, 13, 14, 15);

    int ko = kb >> 3;             // kb in bf16x8 units
    bf16x8 b0a = pb0[ko], b0b = pb0[ko + 1];
    bf16x8 b1a = pb1[ko], b1b = pb1[ko + 1];
    v16bf fb0 = __builtin_shufflevector(b0a, b0b,
        0, 1, 2, 3, 4, 5, 6, 7, 8, 9, 10, 11, 12, 13, 14, 15);
    v16bf fb1 = __builtin_shufflevector(b1a, b1b,
        0, 1, 2, 3, 4, 5, 6, 7, 8, 9, 10, 11, 12, 13, 14, 15);

    acc0 = __builtin_amdgcn_wmma_f32_16x16x32_bf16(
        false, fa, false, fb0, (short)0, acc0, false, false);
    acc1 = __builtin_amdgcn_wmma_f32_16x16x32_bf16(
        false, fa, false, fb1, (short)0, acc1, false, false);
    __syncthreads();
  }

#pragma unroll
  for (int r = 0; r < 8; ++r) {
    int row  = row0 + m_off + ((lane < 16) ? r : r + 8);
    int colA = col0 + n_off + (lane & 15);
#pragma unroll
    for (int half = 0; half < 2; ++half) {
      int col = colA + half * 16;
      float v = (half == 0) ? acc0[r] : acc1[r];
      if (MODE == 0) {
        if (col < Nvalid) Cout[(size_t)row * ldc + col] = v;
      } else if (MODE == 1) {
        if (col < 2 * FREQ) {
          v += (col < FREQ) ? bias_re[col] : bias_im[col - FREQ];
          Cb16[(size_t)row * IRK + col] = f2b(v);
        } else if (col < IRK) {
          Cb16[(size_t)row * IRK + col] = f2b(0.f);
        }
      } else {
        if (col < Hh) {
          int b = row / Cc;
          int c = row - b * Cc;
          size_t oi = ((size_t)b * Hh + col) * Cc + c;
          outp[oi] = yhat[oi] + v * stdv[row];
        }
      }
    }
  }
}

// ---------------------------------------------------------------------------
// k4: fused filter head + spectral combine -> MIXIN (BC x 608 bf16)
// filt[o] = sum_s p[s]*mhb[s,o] + sum_{s,k} p[s]*corr[k]*mhw[s,k,o]
// ---------------------------------------------------------------------------
__global__ __launch_bounds__(128) void k4_combine(
    const float* __restrict__ SF, const float* __restrict__ YF,
    const float* __restrict__ pC, const float* __restrict__ corr,
    const float* __restrict__ sgn,
    const float* __restrict__ mhw, const float* __restrict__ mhb,
    bf16_t* __restrict__ MIXIN) {
  int bc = blockIdx.x;
  int tid = threadIdx.x;
  __shared__ float sp[Ssz], sc[Ssz], ssg[Ssz], sq[Ssz * Kk];
  if (tid < Ssz) {
    sp[tid]  = pC[bc * Ssz + tid];
    sc[tid]  = corr[bc * Kk + tid];
    ssg[tid] = sgn[bc * Kk + tid];
  }
  __syncthreads();
  if (tid < Ssz * Kk) sq[tid] = sp[tid >> 3] * sc[tid & 7];
  __syncthreads();

  int f = tid;
  if (f < FREQ) {
    float fw[17];
    for (int nf = 0; nf < 17; ++nf) {
      int o = nf * FREQ + f;
      float a = 0.f;
#pragma unroll
      for (int s = 0; s < Ssz; ++s) a += sp[s] * mhb[s * ODIM + o];
#pragma unroll
      for (int s = 0; s < Ssz; ++s) {
#pragma unroll
        for (int k = 0; k < Kk; ++k)
          a += sq[s * Kk + k] * mhw[(size_t)s * (Kk * ODIM) + k * ODIM + o];
      }
      fw[nf] = a;
    }
    float yr = YF[(size_t)bc * 194 + f];
    float yi = YF[(size_t)bc * 194 + FREQ + f];
    float Are = 0.f, Aim = 0.f, Bre = 0.f, Bim = 0.f, dsum = 0.f;
#pragma unroll
    for (int k = 0; k < Kk; ++k) {
      size_t sfb = ((size_t)bc * Kk + k) * 194;
      float re = SF[sfb + f], im = SF[sfb + FREQ + f];
      float wl = fw[k] * ssg[k];
      float wd = fw[8 + k] * ssg[k];
      Are += wl * re; Aim += wl * im;
      Bre += wd * re; Bim += wd * im;
      dsum += fw[8 + k];
    }
    Bre -= yr * dsum; Bim -= yi * dsum;
    float Cre = yr * fw[16], Cim = yi * fw[16];
    size_t mb = (size_t)bc * MIXK;
    MIXIN[mb + f]                  = f2b(Are);
    MIXIN[mb + FREQ + f]           = f2b(Bre);
    MIXIN[mb + 2 * FREQ + f]       = f2b(Cre);
    MIXIN[mb + 291 + f]            = f2b(Aim);
    MIXIN[mb + 291 + FREQ + f]     = f2b(Bim);
    MIXIN[mb + 291 + 2 * FREQ + f] = f2b(Cim);
  }
  if (tid >= FREQ && tid < FREQ + 26)  // zero padding cols 582..607
    MIXIN[(size_t)bc * MIXK + 582 + (tid - FREQ)] = f2b(0.f);
}

}  // namespace

// ---------------------------------------------------------------------------
extern "C" void kernel_launch(void* const* d_in, const int* in_sizes, int n_in,
                              void* d_out, int out_size, void* d_ws, size_t ws_size,
                              hipStream_t stream) {
  const float* x     = (const float*)d_in[0];
  const float* yhat  = (const float*)d_in[1];
  const float* r_in  = (const float*)d_in[2];
  const float* temp  = (const float*)d_in[3];
  const float* cw    = (const float*)d_in[4];
  const float* bst   = (const float*)d_in[5];
  const float* sbias = (const float*)d_in[6];
  const float* mhw   = (const float*)d_in[7];
  const float* mhb   = (const float*)d_in[8];
  const float* mwr   = (const float*)d_in[9];
  const float* mwi   = (const float*)d_in[10];
  const float* mbr   = (const float*)d_in[11];
  const float* mbi   = (const float*)d_in[12];
  const int* leader  = (const int*)d_in[13];
  const int* shift   = (const int*)d_in[14];
  float* outp = (float*)d_out;

  char* base = (char*)d_ws;
  size_t off = 0;
  auto alloc = [&](size_t bytes) -> char* {
    char* p = base + off;
    off = (off + bytes + 255) & ~(size_t)255;
    return p;
  };
  float*  stdv  = (float*) alloc((size_t)BC * 4);
  float*  pC    = (float*) alloc((size_t)BC * Ssz * 4);
  float*  corr  = (float*) alloc((size_t)BC * Kk * 4);
  float*  sgn   = (float*) alloc((size_t)BC * Kk * 4);
  float*  yn    = (float*) alloc((size_t)BC * Hh * 4);
  bf16_t* ynb   = (bf16_t*)alloc((size_t)BC * Hh * 2);
  float*  xtail = (float*) alloc((size_t)BC * 48 * 4);
  bf16_t* Sm    = (bf16_t*)alloc((size_t)BCK * Hh * 2);
  float*  SF    = (float*) alloc((size_t)BCK * 194 * 4);
  float*  YF    = (float*) alloc((size_t)BC * 194 * 4);
  bf16_t* MIXIN = (bf16_t*)alloc((size_t)BC * MIXK * 2);
  bf16_t* A4    = (bf16_t*)alloc((size_t)BC * IRK * 2);
  bf16_t* DFTB  = (bf16_t*)alloc((size_t)NB * Hh * 2);    // col-major 256 x 192
  bf16_t* MIXB  = (bf16_t*)alloc((size_t)NB * MIXK * 2);  // col-major 256 x 608
  bf16_t* IRB   = (bf16_t*)alloc((size_t)NB * IRK * 2);   // col-major 256 x 224
  (void)ws_size; (void)n_in; (void)in_sizes; (void)out_size;

  // operator matrices (column-major / fragment-ready)
  init_dft<<<dim3(Hh), dim3(NB), 0, stream>>>(DFTB);
  init_mix<<<dim3(MIXK), dim3(NB), 0, stream>>>(MIXB, mwr, mwi);
  init_ir<<<dim3(IRK), dim3(NB), 0, stream>>>(IRB);

  // stats / softmaxes / normalization
  k1_stats<<<dim3((Cc + 255) / 256, Bb), dim3(256), 0, stream>>>(
      x, yhat, r_in, temp, cw, bst, sbias,
      stdv, pC, corr, sgn, yn, ynb, xtail);

  // gather shifted leader sequences
  k2_gather<<<dim3(BCK), dim3(Hh), 0, stream>>>(leader, shift, yn, xtail, Sm);

  // s_f = DFT(Sm): (110336 x 192) x (192 x 194)
  gemm_wmma<0, Hh><<<dim3(BCK / 32, 2), dim3(256), 0, stream>>>(
      Sm, DFTB, SF, nullptr, nullptr, nullptr, nullptr, nullptr, nullptr,
      BCK, Hh, 194, 194);
  // y_f = DFT(yn): (13792 x 192) x (192 x 194)
  gemm_wmma<0, Hh><<<dim3(BC / 32, 2), dim3(256), 0, stream>>>(
      ynb, DFTB, YF, nullptr, nullptr, nullptr, nullptr, nullptr, nullptr,
      BC, Hh, 194, 194);

  // fused filter head + spectral combine
  k4_combine<<<dim3(BC), dim3(128), 0, stream>>>(
      SF, YF, pC, corr, sgn, mhw, mhb, MIXIN);

  // complex mix: (13792 x 582) x (582 x 194) + bias -> bf16
  gemm_wmma<1, MIXK><<<dim3(BC / 32, 2), dim3(256), 0, stream>>>(
      MIXIN, MIXB, nullptr, A4, mbr, mbi, nullptr, nullptr, nullptr,
      BC, MIXK, 0, 194);

  // irfft + epilogue: out = y_hat + irfft(mix_out)*std  (written as (B,H,C))
  gemm_wmma<2, IRK><<<dim3(BC / 32, 2), dim3(256), 0, stream>>>(
      A4, IRB, nullptr, nullptr, nullptr, nullptr, yhat, stdv, outp,
      BC, IRK, 0, Hh);
}
